// MetaGRUCell_34926674051562
// MI455X (gfx1250) — matrix operand order repaired and checked by reading, not compiled, not run on hardware
//
#include <hip/hip_runtime.h>
#include <cstddef>

// MetaGRU hypernetwork + GRU fused kernel for gfx1250 (MI455X).
// One workgroup per node: generate per-node weights into LDS (f16, transposed),
// then run the 12-step GRU with v_wmma_f32_16x16x32_f16, keeping the hidden
// state tile in accumulator-layout registers across timesteps.

#define NN   1024
#define BB   32
#define TT   12
#define PRE  64
#define HID  64
#define DFEAT 32

typedef __attribute__((ext_vector_type(16))) _Float16 v16h;
typedef __attribute__((ext_vector_type(8)))  float    v8f;

union FragAB { v16h v; _Float16 h[16]; unsigned u[8]; };

struct MlpParams { const float *W0, *b0, *W1, *b1, *W2, *b2; };
struct Args {
    const float* feature;   // [1024,32]
    const float* data;      // [1024,32,12,64]
    const float* h0;        // [1024,32,64]
    MlpParams m[8];         // z_w, z_b, r_w, r_b, i2h_w, i2h_b, h2h_w, h2h_b
    float* out;             // [1024,32,12,64]
    float* state_out;       // [1024,32,64]
};

// Fast sigmoid: v_exp_f32 + v_add + v_rcp_f32 (avoid the IEEE div fixup chain).
__device__ __forceinline__ float sigmoidf_(float x) {
    return __builtin_amdgcn_rcpf(1.0f + __expf(-x));
}

// gfx1250 has a native v_tanh_f32 transcendental; use it when the toolchain
// exposes the builtin, else fall back to the fast-sigmoid identity.
__device__ __forceinline__ float tanhf_(float x) {
#if __has_builtin(__builtin_amdgcn_tanhf)
    return __builtin_amdgcn_tanhf(x);
#elif __has_builtin(__builtin_amdgcn_tanh_f32)
    return __builtin_amdgcn_tanh_f32(x);
#else
    return 2.0f * sigmoidf_(2.0f * x) - 1.0f;
#endif
}

// LDS row strides in halves; +4 padding keeps the 16 lanes of a fragment load
// on distinct banks (stride 132h = 264B = 66 banks -> 2-bank lane stride).
constexpr int DS_ST = 132;  // sDS: 32 rows x (128+4)
constexpr int WZ_ST = 132;  // sWz/sWr (transposed [o][k]): 64 rows x (128+4)
constexpr int WI_ST = 68;   // sWi/sWh (transposed [o][k]): 64 rows x (64+4)
constexpr int RH_ST = 68;   // sRH: 32 rows x (64+4)

__global__ __launch_bounds__(256, 1)
void meta_gru_kernel(Args args) {
    const int n     = blockIdx.x;       // node
    const int tid   = threadIdx.x;      // 0..255 (8 wave32s)
    const int lane  = tid & 31;
    const int wid   = tid >> 5;         // 0..7
    const int mt    = wid >> 2;         // row tile 0..1  (batch rows)
    const int nt    = wid & 3;          // col tile 0..3  (hidden cols)
    const int mlane = lane & 15;
    const int hhalf = lane >> 4;        // 0/1: K-half selector (this code's contiguous k order)

    __shared__ __align__(16) _Float16 sWz[64 * WZ_ST];
    __shared__ __align__(16) _Float16 sWr[64 * WZ_ST];
    __shared__ __align__(16) _Float16 sWi[64 * WI_ST];
    __shared__ __align__(16) _Float16 sWh[64 * WI_ST];
    __shared__ __align__(16) _Float16 sDS[32 * DS_ST];   // [x | h] f16
    __shared__ __align__(16) _Float16 sRH[32 * RH_ST];   // r*h  f16
    __shared__ float sFeat[DFEAT];
    __shared__ float sH1[8][16];
    __shared__ float sH2[8][2];
    __shared__ float sBias[4];          // bz, br, bi, bh

    // ---------------- hypernetwork (per-node, once) ----------------
    if (tid < DFEAT) sFeat[tid] = args.feature[n * DFEAT + tid];
    __syncthreads();

    if (tid < 128) {                    // layer 0 of all 8 tiny MLPs
        const int mi = tid >> 4, j = tid & 15;
        const MlpParams& p = args.m[mi];
        float s = p.b0[j];
        #pragma unroll
        for (int i = 0; i < DFEAT; ++i) s += sFeat[i] * p.W0[i * 16 + j];
        sH1[mi][j] = sigmoidf_(s);
    }
    __syncthreads();
    if (tid < 16) {                     // layer 1 of all 8 tiny MLPs
        const int mi = tid >> 1, j = tid & 1;
        const MlpParams& p = args.m[mi];
        float s = p.b1[j];
        #pragma unroll
        for (int i = 0; i < 16; ++i) s += sH1[mi][i] * p.W1[i * 2 + j];
        sH2[mi][j] = sigmoidf_(s);
    }
    __syncthreads();
    if (tid < 4) {                      // scalar bias heads (b-MLPs: mi=1,3,5,7)
        const int mi = tid * 2 + 1;
        const MlpParams& p = args.m[mi];
        sBias[tid] = sigmoidf_(sH2[mi][0] * p.W2[0] + sH2[mi][1] * p.W2[1] + p.b2[0]);
    }

    // Generate weight matrices into LDS, transposed to [o][k] so B-fragment
    // K-pairs are contiguous 32-bit LDS reads.
    auto gen = [&](int mi, _Float16* dst, int st, int cnt) {
        const MlpParams& p = args.m[mi];
        const float a0 = sH2[mi][0], a1 = sH2[mi][1];
        for (int e = tid; e < cnt; e += 256) {
            float w = sigmoidf_(a0 * p.W2[e] + a1 * p.W2[cnt + e] + p.b2[e]);
            const int i = e >> 6, o = e & 63;   // reshape(-1, in, 64)
            dst[o * st + i] = (_Float16)w;
        }
    };
    gen(0, sWz, WZ_ST, 8192);
    gen(2, sWr, WZ_ST, 8192);
    gen(4, sWi, WI_ST, 4096);
    gen(6, sWh, WI_ST, 4096);
    __syncthreads();

    // ---------------- GRU state tile in registers ----------------
    // f32 16x16 C/D layout: VGPR j holds M=j (lanes 0-15) / M=j+8 (lanes 16-31),
    // N = lane&15.
    const int rowbase = mt * 16;
    const int colbase = nt * 16;
    float hreg[8];
    #pragma unroll
    for (int j = 0; j < 8; ++j) {
        const int row = rowbase + j + 8 * hhalf;
        hreg[j] = args.h0[((size_t)n * BB + row) * HID + colbase + mlane];
    }

    // A fragment (16x32 f16): lane holds row M=mlane; K pairs contiguous:
    // VGPR v -> K = 2v + 16*half (matches loadB).
    auto loadA = [&](const _Float16* base, int st, int row0, int k0) {
        FragAB f;
        #pragma unroll
        for (int v = 0; v < 8; ++v) {
            const int kk = 2 * v + 16 * hhalf;
            f.u[v] = *reinterpret_cast<const unsigned*>(base + (row0 + mlane) * st + k0 + kk);
        }
        return f;
    };
    // B fragment (32x16 f16): lanes 0-15 hold K=0..15, lanes 16-31 K=16..31,
    // 2 K-values per VGPR; N = mlane. Source is transposed [o][k] in LDS.
    auto loadB = [&](const _Float16* baseT, int st, int n0, int k0) {
        FragAB f;
        #pragma unroll
        for (int v = 0; v < 8; ++v) {
            const int kk = 2 * v + 16 * hhalf;
            f.u[v] = *reinterpret_cast<const unsigned*>(baseT + (n0 + mlane) * st + k0 + kk);
        }
        return f;
    };

    const float bz = sBias[0], br = sBias[1], bi = sBias[2], bh = sBias[3];

    for (int t = 0; t < TT; ++t) {
        __syncthreads();   // all waves done reading sDS/sRH from step t-1

        // build ds = [x_t | h] in f16
        {
            const int b  = tid >> 3;            // 0..31
            const int p0 = (tid & 7) * 8;       // 0..56
            const float* src = args.data + (((size_t)n * BB + b) * TT + t) * PRE + p0;
            #pragma unroll
            for (int q = 0; q < 8; ++q)
                sDS[b * DS_ST + p0 + q] = (_Float16)src[q];
        }
        #pragma unroll
        for (int j = 0; j < 8; ++j) {
            const int row = rowbase + j + 8 * hhalf;
            sDS[row * DS_ST + PRE + colbase + mlane] = (_Float16)hreg[j];
        }
        __syncthreads();

        // r = sigmoid(ds @ Wr + br); store r*h to sRH
        v8f accR = {0.f, 0.f, 0.f, 0.f, 0.f, 0.f, 0.f, 0.f};
        #pragma unroll
        for (int kc = 0; kc < 4; ++kc) {
            FragAB fa = loadA(sDS, DS_ST, rowbase, kc * 32);
            FragAB fb = loadB(sWr, WZ_ST, colbase, kc * 32);
            accR = __builtin_amdgcn_wmma_f32_16x16x32_f16(
                false, fa.v, false, fb.v, (short)0, accR, false, false);
        }
        #pragma unroll
        for (int j = 0; j < 8; ++j) {
            const float r = sigmoidf_(accR[j] + br);
            const int row = rowbase + j + 8 * hhalf;
            sRH[row * RH_ST + colbase + mlane] = (_Float16)(r * hreg[j]);
        }

        // z = sigmoid(ds @ Wz + bz)  (reads only sDS; no barrier needed yet)
        v8f accZ = {0.f, 0.f, 0.f, 0.f, 0.f, 0.f, 0.f, 0.f};
        #pragma unroll
        for (int kc = 0; kc < 4; ++kc) {
            FragAB fa = loadA(sDS, DS_ST, rowbase, kc * 32);
            FragAB fb = loadB(sWz, WZ_ST, colbase, kc * 32);
            accZ = __builtin_amdgcn_wmma_f32_16x16x32_f16(
                false, fa.v, false, fb.v, (short)0, accZ, false, false);
        }
        __syncthreads();   // sRH visible to all waves

        // cand = tanh(x @ Wi + (r*h) @ Wh + bi + bh)
        v8f accC = {0.f, 0.f, 0.f, 0.f, 0.f, 0.f, 0.f, 0.f};
        #pragma unroll
        for (int kc = 0; kc < 2; ++kc) {
            FragAB fa = loadA(sDS, DS_ST, rowbase, kc * 32);     // x part (K<64)
            FragAB fb = loadB(sWi, WI_ST, colbase, kc * 32);
            accC = __builtin_amdgcn_wmma_f32_16x16x32_f16(
                false, fa.v, false, fb.v, (short)0, accC, false, false);
        }
        #pragma unroll
        for (int kc = 0; kc < 2; ++kc) {
            FragAB fa = loadA(sRH, RH_ST, rowbase, kc * 32);
            FragAB fb = loadB(sWh, WI_ST, colbase, kc * 32);
            accC = __builtin_amdgcn_wmma_f32_16x16x32_f16(
                false, fa.v, false, fb.v, (short)0, accC, false, false);
        }

        // h_new = z*h + (1-z)*cand ; stream to output
        #pragma unroll
        for (int j = 0; j < 8; ++j) {
            const float z    = sigmoidf_(accZ[j] + bz);
            const float cand = tanhf_(accC[j] + bi + bh);
            const float hn   = z * hreg[j] + (1.0f - z) * cand;
            hreg[j] = hn;
            const int row = rowbase + j + 8 * hhalf;
            args.out[(((size_t)n * BB + row) * TT + t) * HID + colbase + mlane] = hn;
        }
    }

    // final state
    #pragma unroll
    for (int j = 0; j < 8; ++j) {
        const int row = rowbase + j + 8 * hhalf;
        args.state_out[((size_t)n * BB + row) * HID + colbase + mlane] = hreg[j];
    }
}

extern "C" void kernel_launch(void* const* d_in, const int* in_sizes, int n_in,
                              void* d_out, int out_size, void* d_ws, size_t ws_size,
                              hipStream_t stream) {
    (void)in_sizes; (void)n_in; (void)out_size; (void)d_ws; (void)ws_size;
    Args a;
    a.feature = (const float*)d_in[0];
    a.data    = (const float*)d_in[1];
    a.h0      = (const float*)d_in[2];
    // dict order: z_w, z_b, r_w, r_b, i2h_w, i2h_b, h2h_w, h2h_b;
    // each MLP flattens as W0,b0,W1,b1,W2,b2.
    for (int g = 0; g < 8; ++g) {
        const int base = 3 + 6 * g;
        a.m[g].W0 = (const float*)d_in[base + 0];
        a.m[g].b0 = (const float*)d_in[base + 1];
        a.m[g].W1 = (const float*)d_in[base + 2];
        a.m[g].b1 = (const float*)d_in[base + 3];
        a.m[g].W2 = (const float*)d_in[base + 4];
        a.m[g].b2 = (const float*)d_in[base + 5];
    }
    a.out       = (float*)d_out;
    a.state_out = (float*)d_out + (size_t)NN * BB * TT * HID;

    meta_gru_kernel<<<NN, 256, 0, stream>>>(a);
}
